// Net_22625887715641
// MI455X (gfx1250) — compile-verified, matrix operand order checked by reading, not run on hardware
//
#include <hip/hip_runtime.h>

typedef _Float16 f16;
typedef _Float16 __attribute__((ext_vector_type(16))) v16h;
typedef _Float16 __attribute__((ext_vector_type(8)))  v8h;
typedef float    __attribute__((ext_vector_type(8)))  v8f;

// ---------------- geometry ----------------
#define IN_W     394
#define OUT_W    384           // conv 'VALID' output H=W
#define CROP     11            // corr_x == corr_y == 11
#define CW       362           // cropped correlation-template size
#define A_STRIDE 384           // A row stride in halfs (362 valid + zero pad)
#define B_STRIDE 416           // B row stride in halfs (384 valid + zero pad, covers dx+x0+31)
#define NF       32            // total feature channels
#define NSH      23            // 2*11+1 shifts per axis
#define SGRP     4             // dx shifts per workgroup (A-fragment reuse factor)
#define X1N      (NF*OUT_W*OUT_W)       // 4,718,592 floats of x1 output
#define A_CH     (CW*A_STRIDE)          // halfs per A channel plane
#define B_CH     (OUT_W*B_STRIDE)       // halfs per B channel plane
#define A_ELEMS  (NF*A_CH)
#define B_ELEMS  (NF*B_CH)

// ---------------- workspace zero-init (covers all zero padding) ----------------
__global__ void zero_ws_kernel(uint4* __restrict__ p, int n) {
  uint4 z = {0u,0u,0u,0u};
  for (int i = blockIdx.x*blockDim.x + threadIdx.x; i < n; i += gridDim.x*blockDim.x)
    p[i] = z;
}

// ---------------- stage 1: conv3d + relu + channel-normalize + f16 pack ----------------
// block 16x16 output pixels, grid (24,24,2); z==0 -> current frame x (writes x1 + A),
// z==1 -> x_prev (writes B in two shift parities).
__global__ __launch_bounds__(256)
void feats_kernel(const float* __restrict__ x, const float* __restrict__ xp,
                  const float* __restrict__ wt, const float* __restrict__ wn,
                  float* __restrict__ x1out,
                  f16* __restrict__ A, f16* __restrict__ Be, f16* __restrict__ Bo)
{
  __shared__ float s_in[3][26][26];     // 16+10 halo, 3 frames
  __shared__ float s_wt[16*363];        // [16,3,11,11]
  __shared__ float s_wn[16*121];        // [16,1,11,11]

  const int tx = threadIdx.x, ty = threadIdx.y;
  const int tid = ty*16 + tx;
  const int bx = blockIdx.x, by = blockIdx.y, z = blockIdx.z;
  const float* __restrict__ in = (z == 0) ? x : xp;

  for (int i = tid; i < 16*363; i += 256) s_wt[i] = wt[i];
  for (int i = tid; i < 16*121; i += 256) s_wn[i] = wn[i];
  for (int i = tid; i < 3*26*26; i += 256) {
    int t = i/676, r = (i%676)/26, c = i%26;
    s_in[t][r][c] = in[t*(IN_W*IN_W) + (by*16 + r)*IN_W + (bx*16 + c)];
  }
  __syncthreads();

  float acc[32];
  #pragma unroll 1
  for (int c = 0; c < 16; ++c) {        // temporal filters (sum over 3 frames), relu, /2
    float s = 0.f;
    for (int t = 0; t < 3; ++t)
      for (int ky = 0; ky < 11; ++ky) {
        #pragma unroll
        for (int kx = 0; kx < 11; ++kx)
          s += s_in[t][ty+ky][tx+kx] * s_wt[c*363 + t*121 + ky*11 + kx];
      }
    acc[c] = fmaxf(s, 0.f) * 0.5f;
  }
  #pragma unroll 1
  for (int c = 0; c < 16; ++c) {        // non-temporal filters on last frame, relu
    float s = 0.f;
    for (int ky = 0; ky < 11; ++ky) {
      #pragma unroll
      for (int kx = 0; kx < 11; ++kx)
        s += s_in[2][ty+ky][tx+kx] * s_wn[c*121 + ky*11 + kx];
    }
    acc[16+c] = fmaxf(s, 0.f);
  }

  float sum = 0.f;
  for (int c = 0; c < 32; ++c) sum += acc[c];
  const float inv = 1.f / (sum + 2.220446049250313e-16f);   // + np.finfo(float).eps

  const int oy = by*16 + ty, ox = bx*16 + tx;
  for (int c = 0; c < 32; ++c) {
    float v = acc[c] * inv;
    if (z == 0) {
      x1out[c*(OUT_W*OUT_W) + oy*OUT_W + ox] = v;           // x1 (f32 output)
      if (oy >= CROP && oy < CROP+CW && ox >= CROP && ox < CROP+CW)
        A[(size_t)c*A_CH + (oy-CROP)*A_STRIDE + (ox-CROP)] = (f16)v;
    } else {
      Be[(size_t)c*B_CH + oy*B_STRIDE + ox] = (f16)v;       // even-dx copy
      if (ox >= 1)
        Bo[(size_t)c*B_CH + oy*B_STRIDE + (ox-1)] = (f16)v; // odd-dx copy (shift by 1)
    }
  }
}

// ---------------- stage 2: 32x32xK GEMM for 4 shifts via WMMA, shared A ----------------
union Frag { v16h v; v8h h[2]; };

__global__ __launch_bounds__(256)
void corr_kernel(const f16* __restrict__ A, const f16* __restrict__ Be,
                 const f16* __restrict__ Bo, float* __restrict__ out)
{
  const int dxBase = blockIdx.x * SGRP;              // 0,4,8,12,16,20
  const int dy = blockIdx.y;                         // shift row in [0,23)
  const int lane = threadIdx.x & 31;
  const int wave = threadIdx.x >> 5;
  const int m  = lane & 15;                          // row (A) / column (B) within tile
  const int kb = (lane >> 4) << 3;                   // K half-offset: 0 or 8

  // A fragment base (shift-independent; second i-tile at +16*A_CH immediate)
  const f16* a0p = A + (size_t)m*A_CH + kb;

  // per-shift B fragment bases; parity copy keeps every 16B load >=4B aligned
  const f16* bp[SGRP];
  #pragma unroll
  for (int s = 0; s < SGRP; ++s) {
    const int dx  = dxBase + s;                      // dx==23 computed, never written
    const f16* Bsel = (dx & 1) ? Bo : Be;
    const int dxo = dx & ~1;
    bp[s] = Bsel + (size_t)m*B_CH + (size_t)dy*B_STRIDE + dxo + kb;
  }

  v8f acc[SGRP][4];
  #pragma unroll
  for (int s = 0; s < SGRP; ++s)
    #pragma unroll
    for (int t = 0; t < 4; ++t)
      acc[s][t] = (v8f){};

  for (int y = wave; y < CW; y += 8) {               // waves split the K row range
    const f16* ar0 = a0p + (size_t)y*A_STRIDE;
    __builtin_prefetch(ar0 + 8*A_STRIDE, 0, 0);      // global_prefetch_b8: next row group
    __builtin_prefetch(bp[0] + (size_t)(y+8)*B_STRIDE, 0, 0);
    __builtin_prefetch(bp[1] + (size_t)(y+8)*B_STRIDE, 0, 0);
    #pragma unroll 1
    for (int xc = 0; xc < 12; ++xc) {                // 12 * 32 halfs covers 362 (+zero pad)
      const int x0 = xc * 32;
      Frag a0, a1;
      a0.h[0] = *(const v8h*)(ar0 + x0);
      a0.h[1] = *(const v8h*)(ar0 + x0 + 16);
      a1.h[0] = *(const v8h*)(ar0 + 16*A_CH + x0);
      a1.h[1] = *(const v8h*)(ar0 + 16*A_CH + x0 + 16);
      #pragma unroll
      for (int s = 0; s < SGRP; ++s) {
        const f16* br = bp[s] + (size_t)y*B_STRIDE + x0;
        Frag b0, b1;
        b0.h[0] = *(const v8h*)(br);
        b0.h[1] = *(const v8h*)(br + 16);
        b1.h[0] = *(const v8h*)(br + 16*B_CH);
        b1.h[1] = *(const v8h*)(br + 16*B_CH + 16);
        acc[s][0] = __builtin_amdgcn_wmma_f32_16x16x32_f16(false, a0.v, false, b0.v, (short)0, acc[s][0], false, false);
        acc[s][1] = __builtin_amdgcn_wmma_f32_16x16x32_f16(false, a0.v, false, b1.v, (short)0, acc[s][1], false, false);
        acc[s][2] = __builtin_amdgcn_wmma_f32_16x16x32_f16(false, a1.v, false, b0.v, (short)0, acc[s][2], false, false);
        acc[s][3] = __builtin_amdgcn_wmma_f32_16x16x32_f16(false, a1.v, false, b1.v, (short)0, acc[s][3], false, false);
      }
    }
  }

  // cross-wave reduction, one 32KB buffer reused over the 4 shifts
  __shared__ float red[8][4][8][32];
  const float scale = 1.0f / ((float)CW * (float)CW);
  #pragma unroll
  for (int s = 0; s < SGRP; ++s) {
    if (s) __syncthreads();                          // previous pass fully read
    #pragma unroll
    for (int v = 0; v < 8; ++v) {
      red[wave][0][v][lane] = acc[s][0][v];
      red[wave][1][v][lane] = acc[s][1][v];
      red[wave][2][v][lane] = acc[s][2][v];
      red[wave][3][v][lane] = acc[s][3][v];
    }
    __syncthreads();
    const int dx = dxBase + s;
    if (dx < NSH) {
      for (int o = threadIdx.x; o < 1024; o += 256) {  // 4 tiles * 8 vgprs * 32 lanes
        int t = o >> 8, v = (o >> 5) & 7, ln = o & 31;
        float sum = 0.f;
        #pragma unroll
        for (int w = 0; w < 8; ++w) sum += red[w][t][v][ln];
        // C/D 16x16 f32 layout: M = v + 8*(lane>=16), N = lane&15
        int i = ((t >> 1) << 4) + v + ((ln >> 4) << 3);
        int j = ((t & 1)  << 4) + (ln & 15);
        out[X1N + ((i*NF + j)*NSH + dy)*NSH + dx] = sum * scale;
      }
    }
  }
}

// ---------------- launcher ----------------
extern "C" void kernel_launch(void* const* d_in, const int* in_sizes, int n_in,
                              void* d_out, int out_size, void* d_ws, size_t ws_size,
                              hipStream_t stream)
{
  const float* x  = (const float*)d_in[0];
  const float* xp = (const float*)d_in[1];
  const float* wt = (const float*)d_in[2];   // filts_temp  [16,1,3,11,11]
  const float* wn = (const float*)d_in[3];   // filts_notemp[16,1,1,11,11]
  float* out = (float*)d_out;

  char* ws = (char*)d_ws;
  f16* A  = (f16*)(ws);
  f16* Be = (f16*)(ws + (size_t)A_ELEMS * sizeof(f16));
  f16* Bo = (f16*)(ws + ((size_t)A_ELEMS + (size_t)B_ELEMS) * sizeof(f16));

  const size_t totalBytes = ((size_t)A_ELEMS + 2u*(size_t)B_ELEMS) * sizeof(f16); // ~28 MB
  const int n16 = (int)(totalBytes / 16);
  zero_ws_kernel<<<2048, 256, 0, stream>>>((uint4*)d_ws, n16);

  feats_kernel<<<dim3(24,24,2), dim3(16,16), 0, stream>>>(x, xp, wt, wn, out, A, Be, Bo);

  corr_kernel<<<dim3((NSH + SGRP - 1)/SGRP, NSH), 256, 0, stream>>>(A, Be, Bo, out);
}
